// EncoderLayer_69965017252062
// MI455X (gfx1250) — matrix operand order repaired and checked by reading, not compiled, not run on hardware
//
#include <hip/hip_runtime.h>

// ---------------------------------------------------------------------------
// Problem constants (from reference)
// ---------------------------------------------------------------------------
#define B_    2
#define S_    2048
#define N_    4096          // B_*S_
#define H_    1024
#define NH_   16
#define DH_   64
#define T_    64
#define TS_   256
#define BITS_ 8
#define K_    8
#define INV_SCALE 0.35355339059327373f   // 1 / 64^(1/4)

// ---------------------------------------------------------------------------
// Vector types
// ---------------------------------------------------------------------------
typedef __attribute__((ext_vector_type(16))) __bf16 v16bf;
typedef __attribute__((ext_vector_type(8)))  __bf16 v8bf;
typedef __attribute__((ext_vector_type(8)))  float  v8f;
typedef unsigned int u32x4 __attribute__((ext_vector_type(4)));
typedef int          i32x4 __attribute__((ext_vector_type(4)));
typedef int          i32x8 __attribute__((ext_vector_type(8)));

union BF16x16 { v16bf v; v8bf h[2]; };

#define AS3 __attribute__((address_space(3)))

// ---------------------------------------------------------------------------
// CDNA5 data-movement feature detection (device pass only)
// ---------------------------------------------------------------------------
#if defined(__HIP_DEVICE_COMPILE__) && __has_builtin(__builtin_amdgcn_tensor_load_to_lds)
#define HAVE_TDM 1
#else
#define HAVE_TDM 0
#endif
#if defined(__HIP_DEVICE_COMPILE__) && __has_builtin(__builtin_amdgcn_global_load_async_to_lds_b128)
#define HAVE_ASYNC 1
#else
#define HAVE_ASYNC 0
#endif

__device__ __forceinline__ v8f wmma_bf16(const v16bf& a, const v16bf& b, const v8f& c) {
  return __builtin_amdgcn_wmma_f32_16x16x32_bf16(false, a, false, b, (short)0, c, false, false);
}

__device__ __forceinline__ unsigned lds_addr_of(void* p) {
  return (unsigned)(unsigned long long)(AS3 void*)p;
}

#if HAVE_TDM
// One TDM descriptor: 2D tile (tileW x tileH elements of bf16), row-major,
// global row stride = strideElems, LDS padding per pad codes (DW units).
__device__ __forceinline__ void tdm_load_2d(const void* gsrc, unsigned ldsOff,
                                            unsigned tileW, unsigned tileH,
                                            unsigned strideElems,
                                            unsigned padIntCode, unsigned padAmtCode) {
  const unsigned long long ga = (unsigned long long)gsrc;
  u32x4 g0;
  g0[0] = 1u;                                           // count=1, user mode
  g0[1] = ldsOff;                                       // lds_addr
  g0[2] = (unsigned)ga;                                 // global_addr[31:0]
  g0[3] = ((unsigned)(ga >> 32) & 0x01FFFFFFu) | (2u << 30);  // addr[56:32], type=2
  i32x8 g1;
  g1[0] = (int)((1u << 16)                               // data_size = 2 bytes
               | (1u << 20)                              // pad_enable
               | (padIntCode << 22) | (padAmtCode << 25));
  g1[1] = (int)((tileW & 0xFFFFu) << 16);                // tensor_dim0[15:0]
  g1[2] = (int)((tileW >> 16) | ((tileH & 0xFFFFu) << 16)); // dim0 hi | dim1 lo
  g1[3] = (int)((tileH >> 16) | ((tileW & 0xFFFFu) << 16)); // dim1 hi | tile_dim0
  g1[4] = (int)(tileH & 0xFFFFu);                        // tile_dim1 (tile_dim2=0)
  g1[5] = (int)strideElems;                              // tensor_dim0_stride[31:0]
  g1[6] = 0;
  g1[7] = 0;
  const i32x4 z4 = {0, 0, 0, 0};
#if defined(__clang_major__) && __clang_major__ >= 23
  const i32x8 z8 = {0, 0, 0, 0, 0, 0, 0, 0};
  __builtin_amdgcn_tensor_load_to_lds(g0, g1, z4, z4, z8, 0);
#else
  __builtin_amdgcn_tensor_load_to_lds(g0, g1, z4, z4, 0);
#endif
}
__device__ __forceinline__ void wait_stage_cnt() {
#if __has_builtin(__builtin_amdgcn_s_wait_tensorcnt)
  __builtin_amdgcn_s_wait_tensorcnt(0);
#else
  asm volatile("s_wait_tensorcnt 0x0" ::: "memory");
#endif
}
#elif HAVE_ASYNC
__device__ __forceinline__ void wait_stage_cnt() {
#if __has_builtin(__builtin_amdgcn_s_wait_asynccnt)
  __builtin_amdgcn_s_wait_asynccnt(0);
#else
  asm volatile("s_wait_asynccnt 0x0" ::: "memory");
#endif
}
#else
__device__ __forceinline__ void wait_stage_cnt() {}
#endif

// Per-lane 16-byte global -> LDS copy (async where available)
__device__ __forceinline__ void copy_b128(const __bf16* g, __bf16* l) {
#if HAVE_ASYNC
  __builtin_amdgcn_global_load_async_to_lds_b128((i32x4*)(void*)g, (i32x4*)(void*)l, 0, 0);
#else
  *(v8bf*)l = *(const v8bf*)g;
#endif
}

// Stage a (ROWS x COLS) bf16 tile (ROWS*COLS == 2048) with 128 threads.
template <int COLS, int LDL>
__device__ __forceinline__ void stage_tile(const __bf16* g, int ldg, __bf16* l, int tid) {
  const int e   = tid * 16;
  const int row = e / COLS;
  const int col = e % COLS;
  copy_b128(g + (size_t)row * ldg + col,     l + row * LDL + col);
  copy_b128(g + (size_t)row * ldg + col + 8, l + row * LDL + col + 8);
}

// ---------------------------------------------------------------------------
// f32 -> bf16 transpose-convert: dst[n][k] = src[k][n]  (H_ x H_)
// ---------------------------------------------------------------------------
__global__ __launch_bounds__(256)
void cvt_transpose_kernel(const float* __restrict__ src, __bf16* __restrict__ dst) {
  __shared__ __bf16 t[32][33];
  const int k0 = blockIdx.x * 32;
  const int n0 = blockIdx.y * 32;
  const int tx = threadIdx.x & 31;
  const int ty = threadIdx.x >> 5;     // 0..7
#pragma unroll
  for (int j = 0; j < 32; j += 8)
    t[ty + j][tx] = (__bf16)src[(size_t)(k0 + ty + j) * H_ + n0 + tx];
  __syncthreads();
#pragma unroll
  for (int j = 0; j < 32; j += 8)
    dst[(size_t)(n0 + ty + j) * H_ + k0 + tx] = t[tx][ty + j];
}

// ---------------------------------------------------------------------------
// LayerNorm over H_=1024, one block (256 threads) per row, bf16 output
// ---------------------------------------------------------------------------
__global__ __launch_bounds__(256)
void ln_bf16_kernel(const float* __restrict__ x, const float* __restrict__ g,
                    const float* __restrict__ b, __bf16* __restrict__ out) {
  __shared__ float r1[256];
  __shared__ float r2[256];
  const int row = blockIdx.x;
  const int tid = threadIdx.x;
  const float* xr = x + (size_t)row * H_;
  float v[4];
  float s = 0.f, ss = 0.f;
#pragma unroll
  for (int j = 0; j < 4; ++j) {
    v[j] = xr[tid + 256 * j];
    s += v[j];
    ss += v[j] * v[j];
  }
  r1[tid] = s; r2[tid] = ss;
  __syncthreads();
  for (int st = 128; st > 0; st >>= 1) {
    if (tid < st) { r1[tid] += r1[tid + st]; r2[tid] += r2[tid + st]; }
    __syncthreads();
  }
  const float mu   = r1[0] * (1.0f / H_);
  const float var  = r2[0] * (1.0f / H_) - mu * mu;
  const float rstd = rsqrtf(var + 1e-12f);
#pragma unroll
  for (int j = 0; j < 4; ++j) {
    const int c = tid + 256 * j;
    out[(size_t)row * H_ + c] = (__bf16)((v[j] - mu) * rstd * g[c] + b[c]);
  }
}

// ---------------------------------------------------------------------------
// bf16 WMMA GEMM: C[64x64 tile] = A[N x K] * Wt[Ncols x K]^T
//   Wt is the pre-transposed weight (row n = output column), so both LDS
//   tiles are row-major 64x32 copies -> TDM/async friendly.
//   Double-buffered LDS: next tiles DMA while current tiles feed WMMA.
// ---------------------------------------------------------------------------
#define GLDA 40   // LDS row stride (halves): 32 data + 8 pad (=16DW + 4DW pad)

__global__ __launch_bounds__(128)
void gemm_bf16_kernel(const __bf16* __restrict__ A, int lda,
                      const __bf16* __restrict__ Wt, int ldwt,
                      const float* __restrict__ bias,
                      const float* __restrict__ residual,
                      float* __restrict__ outF,
                      __bf16* __restrict__ outB,
                      int ldc, float bfScale, int Kdim) {
  __shared__ __bf16 sA[2][64 * GLDA];
  __shared__ __bf16 sW[2][64 * GLDA];

  const int tid  = threadIdx.x;
  const int wid  = tid >> 5;
  const int lane = tid & 31;
  const int half = lane >> 4;
  const int lm   = lane & 15;
  const int r0   = blockIdx.x * 64;
  const int n0   = blockIdx.y * 64;

  v8f acc[4] = {};

  auto stage = [&](int buf, int k0) {
#if HAVE_TDM
    if (wid == 0) {   // one wave issues both tile DMAs (EXEC-independent)
      tdm_load_2d(A  + (size_t)r0 * lda  + k0, lds_addr_of(&sA[buf][0]),
                  32u, 64u, (unsigned)lda,  3u, 3u);   // 16DW rows, 4DW pad
      tdm_load_2d(Wt + (size_t)n0 * ldwt + k0, lds_addr_of(&sW[buf][0]),
                  32u, 64u, (unsigned)ldwt, 3u, 3u);
    }
#else
    stage_tile<32, GLDA>(A  + (size_t)r0 * lda  + k0, lda,  &sA[buf][0], tid);
    stage_tile<32, GLDA>(Wt + (size_t)n0 * ldwt + k0, ldwt, &sW[buf][0], tid);
#endif
  };

  stage(0, 0);
  wait_stage_cnt();
  __syncthreads();

  for (int k0 = 0; k0 < Kdim; k0 += 32) {
    const int cur = (k0 >> 5) & 1;
    if (k0 + 32 < Kdim) stage(cur ^ 1, k0 + 32);   // overlap next DMA with WMMA

    // ---- fragments (ISA 16-bit 16x32 A layout / B layout) ----
    BF16x16 af, bfr[4];
    {
      const __bf16* ap = &sA[cur][(wid * 16 + lm) * GLDA];
      af.h[0] = *(const v8bf*)(ap + half * 8);
      af.h[1] = *(const v8bf*)(ap + 16 + half * 8);
    }
#pragma unroll
    for (int nt = 0; nt < 4; ++nt) {
      const __bf16* bp = &sW[cur][(nt * 16 + lm) * GLDA + half * 8];
      bfr[nt].h[0] = *(const v8bf*)(bp);
      bfr[nt].h[1] = *(const v8bf*)(bp + 16);
    }
#pragma unroll
    for (int nt = 0; nt < 4; ++nt)
      acc[nt] = wmma_bf16(af.v, bfr[nt].v, acc[nt]);

    wait_stage_cnt();
    __syncthreads();
  }

  // ---- epilogue: +bias, +optional residual, f32 and/or scaled bf16 out ----
#pragma unroll
  for (int nt = 0; nt < 4; ++nt) {
    const int col = n0 + nt * 16 + lm;
    const float bb = bias ? bias[col] : 0.0f;
#pragma unroll
    for (int r = 0; r < 8; ++r) {
      const int row = r0 + wid * 16 + half * 8 + r;
      float v = acc[nt][r] + bb;
      if (residual) v += residual[(size_t)row * ldc + col];
      if (outF) outF[(size_t)row * ldc + col] = v;
      if (outB) outB[(size_t)row * ldc + col] = (__bf16)(v * bfScale);
    }
  }
}

// ---------------------------------------------------------------------------
// Flash attention: block = 4 waves -> 64 queries of one (b,h).
//   K tile (32x64) staged by TDM/async; V^T staged manually (transpose).
// ---------------------------------------------------------------------------
#define ALDK 72   // 64 data + 8 pad halves (=32DW + 4DW pad)
#define ALDV 40
#define ALDP 40

__global__ __launch_bounds__(128)
void attn_kernel(const __bf16* __restrict__ Q, const __bf16* __restrict__ Km,
                 const __bf16* __restrict__ Vm, const float* __restrict__ mask,
                 __bf16* __restrict__ ctx) {
  __shared__ __bf16 sK[32 * ALDK];
  __shared__ __bf16 sVt[64 * ALDV];
  __shared__ __bf16 sP[4][16 * ALDP];

  const int b  = blockIdx.x >> 4;   // NH_=16
  const int h  = blockIdx.x & 15;
  const int q0 = blockIdx.y * 64;
  const int tid  = threadIdx.x;
  const int wid  = tid >> 5;
  const int lane = tid & 31;
  const int half = lane >> 4;
  const int lm   = lane & 15;
  const size_t hcol = (size_t)h * DH_;

  // Q fragments (contraction DH_=64 -> two K=32 chunks), kept in registers
  BF16x16 aq[2];
  {
    const int qr = q0 + wid * 16 + lm;
    const __bf16* qp = Q + ((size_t)(b * S_ + qr)) * H_ + hcol;
#pragma unroll
    for (int c = 0; c < 2; ++c) {
      aq[c].h[0] = *(const v8bf*)(qp + c * 32 + half * 8);
      aq[c].h[1] = *(const v8bf*)(qp + c * 32 + 16 + half * 8);
    }
  }

  float mrow[8], lrow[8];
#pragma unroll
  for (int r = 0; r < 8; ++r) { mrow[r] = -1e30f; lrow[r] = 0.0f; }
  v8f o[4] = {};

  const int skey = tid >> 2;         // 0..31 key row in tile
  const int sdof = (tid & 3) * 16;   // feature offset

  for (int kb = 0; kb < S_; kb += 32) {
    const __bf16* kbase = Km + ((size_t)(b * S_ + kb)) * H_ + hcol;
    // ---- K tile via DMA ----
#if HAVE_TDM
    if (wid == 0)
      tdm_load_2d(kbase, lds_addr_of(sK), 64u, 32u, (unsigned)H_, 4u, 3u);
#else
    stage_tile<64, ALDK>(kbase, H_, sK, tid);
#endif
    // ---- V^T tile, manual transpose (overlaps the K DMA) ----
    {
      const __bf16* vs = Vm + ((size_t)(b * S_ + kb + skey)) * H_ + hcol + sdof;
      v8bf v0v = *(const v8bf*)vs;
      v8bf v1v = *(const v8bf*)(vs + 8);
#pragma unroll
      for (int j = 0; j < 8; ++j) sVt[(sdof + j) * ALDV + skey]     = v0v[j];
#pragma unroll
      for (int j = 0; j < 8; ++j) sVt[(sdof + 8 + j) * ALDV + skey] = v1v[j];
    }
    wait_stage_cnt();
    __syncthreads();

    // ---- S = Q K^T for 2 n-tiles of 16 keys ----
    v8f s[2] = {};
#pragma unroll
    for (int nt = 0; nt < 2; ++nt) {
#pragma unroll
      for (int c = 0; c < 2; ++c) {
        BF16x16 bk;
        const __bf16* kp = &sK[(nt * 16 + lm) * ALDK + c * 32 + half * 8];
        bk.h[0] = *(const v8bf*)kp;
        bk.h[1] = *(const v8bf*)(kp + 16);
        s[nt] = wmma_bf16(aq[c].v, bk.v, s[nt]);
      }
      const float madd =
          -1000.0f * (1.0f - mask[(size_t)b * S_ + kb + nt * 16 + lm]);
#pragma unroll
      for (int r = 0; r < 8; ++r) s[nt][r] += madd;
    }

    // ---- online softmax (row-wise over the 32 keys of this block) ----
#pragma unroll
    for (int r = 0; r < 8; ++r) {
      float v0 = s[0][r], v1 = s[1][r];
      float mx = fmaxf(v0, v1);
#pragma unroll
      for (int d = 1; d < 16; d <<= 1) mx = fmaxf(mx, __shfl_xor(mx, d, 32));
      const float mnew = fmaxf(mrow[r], mx);
      const float corr = __expf(mrow[r] - mnew);
      const float p0 = __expf(v0 - mnew);
      const float p1 = __expf(v1 - mnew);
      float rs = p0 + p1;
#pragma unroll
      for (int d = 1; d < 16; d <<= 1) rs += __shfl_xor(rs, d, 32);
      lrow[r] = lrow[r] * corr + rs;
      mrow[r] = mnew;
#pragma unroll
      for (int nt = 0; nt < 4; ++nt) o[nt][r] *= corr;
      // spill P (D-frag layout) to LDS to re-fetch as A-frag (same-wave DS order)
      sP[wid][(half * 8 + r) * ALDP + lm]      = (__bf16)p0;
      sP[wid][(half * 8 + r) * ALDP + 16 + lm] = (__bf16)p1;
    }

    // ---- O += P V ----
    BF16x16 ap;
    {
      const __bf16* pp = &sP[wid][lm * ALDP];
      ap.h[0] = *(const v8bf*)(pp + half * 8);
      ap.h[1] = *(const v8bf*)(pp + 16 + half * 8);
    }
#pragma unroll
    for (int nt = 0; nt < 4; ++nt) {
      BF16x16 bv;
      const __bf16* vp = &sVt[(nt * 16 + lm) * ALDV + half * 8];
      bv.h[0] = *(const v8bf*)vp;
      bv.h[1] = *(const v8bf*)(vp + 16);
      o[nt] = wmma_bf16(ap.v, bv.v, o[nt]);
    }
    __syncthreads();
  }

  // ---- normalize + store ctx (bf16, [token][H] layout) ----
#pragma unroll
  for (int nt = 0; nt < 4; ++nt) {
#pragma unroll
    for (int r = 0; r < 8; ++r) {
      const int row = q0 + wid * 16 + half * 8 + r;
      ctx[((size_t)(b * S_ + row)) * H_ + hcol + nt * 16 + lm] =
          (__bf16)(o[nt][r] / lrow[r]);
    }
  }
}

// ---------------------------------------------------------------------------
// Per (token, table): 256 sign-dot logits, top-8 (first-occurrence ties),
// softmax -> weights + indices.  logit(c) = 2*sum_{bit set} h_b - sum h_b.
// ---------------------------------------------------------------------------
__global__ __launch_bounds__(256)
void topk_kernel(const float* __restrict__ hs, float* __restrict__ wout,
                 int* __restrict__ iout) {
  const int gid = blockIdx.x * 256 + threadIdx.x;
  if (gid >= N_ * T_) return;
  const int n = gid / T_;
  const int t = gid % T_;
  const float* hv = hs + (size_t)n * (T_ * BITS_) + t * BITS_;
  float hval[8];
  float tot = 0.f;
#pragma unroll
  for (int j = 0; j < 8; ++j) { hval[j] = hv[j]; tot += hval[j]; }

  float val[8]; int idx[8];
#pragma unroll
  for (int j = 0; j < 8; ++j) { val[j] = -3.0e38f; idx[j] = 0; }

  for (int c = 0; c < TS_; ++c) {
    float a = 0.f;
#pragma unroll
    for (int j = 0; j < 8; ++j) a += ((c >> j) & 1) ? hval[j] : 0.0f;
    const float lg = 2.0f * a - tot;
    if (lg > val[7]) {
      int j = 7;
      while (j > 0 && lg > val[j - 1]) { val[j] = val[j - 1]; idx[j] = idx[j - 1]; --j; }
      val[j] = lg; idx[j] = c;
    }
  }

  const float mx = val[0];
  float e[8];
  float se = 0.f;
#pragma unroll
  for (int j = 0; j < 8; ++j) { e[j] = __expf(val[j] - mx); se += e[j]; }
  const float inv = 1.0f / se;
  float* wo = wout + (size_t)gid * 8;
  int*   io = iout + (size_t)gid * 8;
#pragma unroll
  for (int j = 0; j < 8; ++j) { wo[j] = e[j] * inv; io[j] = idx[j]; }
}

// ---------------------------------------------------------------------------
// Gather/accumulate: out[n] = attn_out[n] + tables_bias + sum_{t,k} w * row
// One block per token; w/idx broadcast via LDS; table rows are L2-resident.
// ---------------------------------------------------------------------------
__global__ __launch_bounds__(256)
void gather_kernel(const float* __restrict__ w, const int* __restrict__ idx,
                   const float* __restrict__ tab, const float* __restrict__ tbias,
                   const float* __restrict__ attn, float* __restrict__ out) {
  __shared__ float sw[T_ * K_];
  __shared__ int   si[T_ * K_];
  const int n = blockIdx.x;
  const int tid = threadIdx.x;
  sw[tid]       = w[(size_t)n * 512 + tid];
  sw[tid + 256] = w[(size_t)n * 512 + tid + 256];
  si[tid]       = idx[(size_t)n * 512 + tid];
  si[tid + 256] = idx[(size_t)n * 512 + tid + 256];
  __syncthreads();

  float acc[4];
#pragma unroll
  for (int j = 0; j < 4; ++j) {
    const int c = tid + 256 * j;
    acc[j] = attn[(size_t)n * H_ + c] + tbias[c];
  }
  for (int i = 0; i < T_ * K_; ++i) {
    const int t = i >> 3;
    const float* trow = tab + ((size_t)(t * TS_) + si[i]) * H_;
    if (i + 1 < T_ * K_) {
      const int tn = (i + 1) >> 3;
      __builtin_prefetch(tab + ((size_t)(tn * TS_) + si[i + 1]) * H_ + tid, 0, 3);
    }
    const float wv = sw[i];
#pragma unroll
    for (int j = 0; j < 4; ++j) acc[j] += wv * trow[tid + 256 * j];
  }
#pragma unroll
  for (int j = 0; j < 4; ++j) out[(size_t)n * H_ + tid + 256 * j] = acc[j];
}

// ---------------------------------------------------------------------------
// Host launch
// ---------------------------------------------------------------------------
extern "C" void kernel_launch(void* const* d_in, const int* in_sizes, int n_in,
                              void* d_out, int out_size, void* d_ws, size_t ws_size,
                              hipStream_t stream) {
  const float* hidden = (const float*)d_in[0];
  const float* amask  = (const float*)d_in[1];
  const float* ln1g   = (const float*)d_in[2];
  const float* ln1b   = (const float*)d_in[3];
  const float* Wq = (const float*)d_in[4];
  const float* bq = (const float*)d_in[5];
  const float* Wk = (const float*)d_in[6];
  const float* bk = (const float*)d_in[7];
  const float* Wv = (const float*)d_in[8];
  const float* bv = (const float*)d_in[9];
  const float* Wo = (const float*)d_in[10];
  const float* bo = (const float*)d_in[11];
  const float* ln2g = (const float*)d_in[12];
  const float* ln2b = (const float*)d_in[13];
  const float* Wh = (const float*)d_in[14];
  const float* bh = (const float*)d_in[15];
  const float* tabw = (const float*)d_in[16];
  const float* tabb = (const float*)d_in[17];
  float* out = (float*)d_out;

  // ---- workspace carve-up (~103 MB) ----
  char* p = (char*)d_ws;
  auto carve = [&](size_t bytes) -> char* {
    char* r = p;
    p += (bytes + 255) & ~(size_t)255;
    return r;
  };
  const size_t NB_ACT = (size_t)N_ * H_ * sizeof(__bf16);   // 8 MB
  const size_t NB_W   = (size_t)H_ * H_ * sizeof(__bf16);   // 2 MB
  __bf16* xln1 = (__bf16*)carve(NB_ACT);
  __bf16* wqt  = (__bf16*)carve(NB_W);    // transposed bf16 weights [n][k]
  __bf16* wkt  = (__bf16*)carve(NB_W);
  __bf16* wvt  = (__bf16*)carve(NB_W);
  __bf16* wot  = (__bf16*)carve(NB_W);
  __bf16* wht  = (__bf16*)carve(NB_W);
  __bf16* qb   = (__bf16*)carve(NB_ACT);
  __bf16* kb2  = (__bf16*)carve(NB_ACT);
  __bf16* vb2  = (__bf16*)carve(NB_ACT);
  __bf16* ctxb = (__bf16*)carve(NB_ACT);
  float*  attn = (float*)carve((size_t)N_ * H_ * sizeof(float));         // 16 MB
  __bf16* yb   = (__bf16*)carve(NB_ACT);
  float*  hsf  = (float*)carve((size_t)N_ * T_ * BITS_ * sizeof(float)); // 8 MB
  float*  wtop = (float*)carve((size_t)N_ * T_ * K_ * sizeof(float));    // 8 MB
  int*    itop = (int*)carve((size_t)N_ * T_ * K_ * sizeof(int));        // 8 MB

  // 1) transpose-convert weights to bf16 [outCol][inDim]
  dim3 gT(H_ / 32, H_ / 32);
  cvt_transpose_kernel<<<gT, 256, 0, stream>>>(Wq, wqt);
  cvt_transpose_kernel<<<gT, 256, 0, stream>>>(Wk, wkt);
  cvt_transpose_kernel<<<gT, 256, 0, stream>>>(Wv, wvt);
  cvt_transpose_kernel<<<gT, 256, 0, stream>>>(Wo, wot);
  cvt_transpose_kernel<<<gT, 256, 0, stream>>>(Wh, wht);

  // 2) LayerNorm 1 -> bf16
  ln_bf16_kernel<<<N_, 256, 0, stream>>>(hidden, ln1g, ln1b, xln1);

  // 3) Q,K,V projections (q,k pre-scaled by 1/sqrt(sqrt(DH)))
  dim3 gFull(N_ / 64, H_ / 64);
  gemm_bf16_kernel<<<gFull, 128, 0, stream>>>(xln1, H_, wqt, H_, bq, nullptr,
                                              nullptr, qb, H_, INV_SCALE, H_);
  gemm_bf16_kernel<<<gFull, 128, 0, stream>>>(xln1, H_, wkt, H_, bk, nullptr,
                                              nullptr, kb2, H_, INV_SCALE, H_);
  gemm_bf16_kernel<<<gFull, 128, 0, stream>>>(xln1, H_, wvt, H_, bv, nullptr,
                                              nullptr, vb2, H_, 1.0f, H_);

  // 4) flash attention -> ctx (bf16)
  attn_kernel<<<dim3(B_ * NH_, S_ / 64), 128, 0, stream>>>(qb, kb2, vb2, amask, ctxb);

  // 5) output projection + residual -> attn_out (f32)
  gemm_bf16_kernel<<<gFull, 128, 0, stream>>>(ctxb, H_, wot, H_, bo, hidden,
                                              attn, nullptr, H_, 1.0f, H_);

  // 6) LayerNorm 2 -> bf16
  ln_bf16_kernel<<<N_, 256, 0, stream>>>(attn, ln2g, ln2b, yb);

  // 7) hs = (y @ Wh + bh)[:, :512]   (rows 0..511 of Wh^T)
  gemm_bf16_kernel<<<dim3(N_ / 64, (T_ * BITS_) / 64), 128, 0, stream>>>(
      yb, H_, wht, H_, bh, nullptr, hsf, nullptr, T_ * BITS_, 1.0f, H_);

  // 8) logits + top-8 + softmax weights
  topk_kernel<<<(N_ * T_) / 256, 256, 0, stream>>>(hsf, wtop, itop);

  // 9) table gather + bias + residual -> out
  gather_kernel<<<N_, 256, 0, stream>>>(wtop, itop, tabw, tabb, attn, out);

  (void)in_sizes; (void)n_in; (void)out_size; (void)ws_size;
}
